// Qwen3NextGatedDeltaNet_85864986182162
// MI455X (gfx1250) — compile-verified, matrix operand order checked by reading, not run on hardware
//
#include <hip/hip_runtime.h>

// ---------------------------------------------------------------- types
typedef __attribute__((ext_vector_type(16))) __bf16    v16bf;
typedef __attribute__((ext_vector_type(8)))  float     v8f;
typedef __attribute__((ext_vector_type(4)))  unsigned  u32x4;

#define SEQ   4096
#define HID   2048
#define HK    16
#define HV    32
#define DK    128
#define DV    128
#define KEYD  2048          // HK*DK
#define VALD  4096          // HV*DV
#define NQKVZ 12288         // 2*KEYD + 2*VALD
#define CONVD 8192          // 2*KEYD + VALD

__device__ __forceinline__ unsigned short f2bf(float x) {
  unsigned u = __float_as_uint(x);
  u += 0x7FFFu + ((u >> 16) & 1u);          // round-to-nearest-even
  return (unsigned short)(u >> 16);
}
__device__ __forceinline__ float bf2f(unsigned short h) {
  return __uint_as_float(((unsigned)h) << 16);
}
__device__ __forceinline__ float sigm(float x) { return 1.f / (1.f + expf(-x)); }

// ---------------------------------------------------------------- fp32 -> bf16
__global__ void k_f32_to_bf16(const float* __restrict__ x,
                              unsigned short* __restrict__ y, int n) {
  int i = blockIdx.x * 256 + threadIdx.x;
  if (i < n) y[i] = f2bf(x[i]);
}

// ---------------------------------------------------------------- tiled transpose
// W [K][N] fp32 (row major) -> Wt [N][K] bf16.  32x32 LDS tiles, coalesced
// on both sides (naive version wastes 32x HBM bandwidth on the strided leg).
// K, N multiples of 32. block = 256 = (32 cols x 8 rows), 4 rows per thread.
__global__ __launch_bounds__(256) void k_transpose_bf16(
    const float* __restrict__ W, unsigned short* __restrict__ Wt, int K, int N) {
  __shared__ float tile[32][33];            // +1 pitch: no LDS bank conflicts
  const int tx = threadIdx.x & 31;
  const int ty = threadIdx.x >> 5;          // 0..7
  const int k0 = blockIdx.y * 32;
  const int n0 = blockIdx.x * 32;
#pragma unroll
  for (int r = 0; r < 4; r++) {
    int k = ty + r * 8;
    tile[k][tx] = W[(size_t)(k0 + k) * N + (n0 + tx)];
  }
  __syncthreads();
#pragma unroll
  for (int r = 0; r < 4; r++) {
    int n = ty + r * 8;
    Wt[(size_t)(n0 + n) * K + (k0 + tx)] = f2bf(tile[tx][n]);
  }
}

// ---------------------------------------------------------------- WMMA GEMM
// C[M][N] = A[M][K] * Bt[N][K]^T, bf16 inputs, f32 accumulate.
// block = 256 thr = 8 waves (2 along M x 4 along N); wave tile 32x64.
union FragBF { v16bf v; u32x4 q[2]; };

__device__ __forceinline__ v16bf load_frag(const unsigned short* __restrict__ row,
                                           int k0) {
  // 16-bit A/B layout: per lane the 16 elements are two contiguous 8-elem runs
  // at K = k0 and K = k0+16  (k0 already includes half*8).
  FragBF f;
  f.q[0] = *(const u32x4*)(row + k0);
  f.q[1] = *(const u32x4*)(row + k0 + 16);
  return f.v;
}

template <bool STORE_BF16>
__global__ __launch_bounds__(256) void k_gemm_bf16(
    const unsigned short* __restrict__ A,   // [M][K] bf16
    const unsigned short* __restrict__ Bt,  // [N][K] bf16 (B transposed)
    void* __restrict__ C, int M, int N, int K) {
  const int lane = threadIdx.x & 31;
  const int wave = threadIdx.x >> 5;
  const int half = lane >> 4;        // 0/1: which 16-lane half
  const int l16  = lane & 15;
  const int wm   = wave & 1;         // 2 waves along M
  const int wn   = wave >> 1;        // 4 waves along N
  const int tile_m = blockIdx.y * 64  + wm * 32;
  const int tile_n = blockIdx.x * 256 + wn * 64;

  v8f acc[2][4];
#pragma unroll
  for (int i = 0; i < 2; i++)
#pragma unroll
    for (int j = 0; j < 4; j++)
#pragma unroll
      for (int e = 0; e < 8; e++) acc[i][j][e] = 0.f;

  const unsigned short* arow[2];
  arow[0] = A + (size_t)(tile_m + l16) * K;
  arow[1] = arow[0] + (size_t)16 * K;
  const unsigned short* brow[4];
#pragma unroll
  for (int j = 0; j < 4; j++) brow[j] = Bt + (size_t)(tile_n + j * 16 + l16) * K;

  for (int k0 = half * 8; k0 < K; k0 += 32) {
    __builtin_prefetch(arow[0] + k0 + 256, 0, 3);
    __builtin_prefetch(arow[1] + k0 + 256, 0, 3);
    v16bf a[2], b[4];
#pragma unroll
    for (int i = 0; i < 2; i++) a[i] = load_frag(arow[i], k0);
#pragma unroll
    for (int j = 0; j < 4; j++) b[j] = load_frag(brow[j], k0);
#pragma unroll
    for (int i = 0; i < 2; i++)
#pragma unroll
      for (int j = 0; j < 4; j++)
        acc[i][j] = __builtin_amdgcn_wmma_f32_16x16x32_bf16(
            false, a[i], false, b[j], (short)0, acc[i][j], false, false);
  }

  // C layout: vgpr r -> m = r + 8*half ; n = lane&15
#pragma unroll
  for (int i = 0; i < 2; i++)
#pragma unroll
    for (int j = 0; j < 4; j++) {
      int n = tile_n + j * 16 + l16;
#pragma unroll
      for (int r = 0; r < 8; r++) {
        int m = tile_m + i * 16 + half * 8 + r;
        if (STORE_BF16)
          ((unsigned short*)C)[(size_t)m * N + n] = f2bf(acc[i][j][r]);
        else
          ((float*)C)[(size_t)m * N + n] = acc[i][j][r];
      }
    }
}

// ---------------------------------------------------------------- b/a -> beta, exp(g)
// ba = hidden @ W_ba ; ba.reshape(L,16,4): b col = (hv>>1)*4+(hv&1), a col = +2
__global__ __launch_bounds__(64) void k_ba(
    const float* __restrict__ hidden, const float* __restrict__ Wba,
    const float* __restrict__ A_log, const float* __restrict__ dt_bias,
    float* __restrict__ gexp, float* __restrict__ beta) {
  int t = blockIdx.x;
  int j = threadIdx.x;            // 0..63
  int hv = j & 31;
  int is_a = j >> 5;
  int col = ((hv >> 1) << 2) + (hv & 1) + (is_a ? 2 : 0);
  const float* h = hidden + (size_t)t * HID;
  float acc = 0.f;
  for (int k = 0; k < HID; k++) acc += h[k] * Wba[(size_t)k * 64 + col];
  if (is_a) {
    float x  = acc + dt_bias[hv];
    float sp = (x > 20.f) ? x : log1pf(expf(x));        // softplus
    gexp[(size_t)t * HV + hv] = expf(-expf(A_log[hv]) * sp);
  } else {
    beta[(size_t)t * HV + hv] = sigm(acc);
  }
}

// ---------------------------------------------------------------- conv + silu + l2norm
// qkvz row layout per key-head h (768 cols): [q 128 | k 128 | v 256 | z 256]
__global__ __launch_bounds__(256) void k_conv(
    const unsigned short* __restrict__ qkvz, const float* __restrict__ conv_w,
    float* __restrict__ qn, float* __restrict__ kn, float* __restrict__ vc) {
  int t = blockIdx.x;
  __shared__ float buf[4096];     // conv+silu for q,k channels
  __shared__ float scale[32];     // per 128-ch group l2 scale

  for (int c = threadIdx.x; c < CONVD; c += 256) {
    int col;
    if (c < KEYD)            col = ((c >> 7) * 768) + (c & 127);
    else if (c < 2 * KEYD) { int cc = c - KEYD;     col = ((cc >> 7) * 768) + 128 + (cc & 127); }
    else                   { int cc = c - 2 * KEYD; col = ((cc >> 8) * 768) + 256 + (cc & 255); }
    float acc = 0.f;
#pragma unroll
    for (int i = 0; i < 4; i++) {
      int tt = t - 3 + i;
      if (tt >= 0)
        acc += bf2f(qkvz[(size_t)tt * NQKVZ + col]) * conv_w[c * 4 + i];
    }
    float s = acc * sigm(acc);    // silu
    if (c < 2 * KEYD) buf[c] = s;
    else              vc[(size_t)t * VALD + (c - 2 * KEYD)] = s;
  }
  __syncthreads();
  if (threadIdx.x < 32) {
    int g = threadIdx.x;          // 16 q-groups then 16 k-groups
    float ss = 0.f;
    for (int d = 0; d < 128; d++) { float x = buf[g * 128 + d]; ss += x * x; }
    float sc = rsqrtf(ss + 1e-6f);
    if (g < 16) sc *= 0.08838834764831845f;   // * DK^-0.5 for q
    scale[g] = sc;
  }
  __syncthreads();
  for (int c = threadIdx.x; c < 2 * KEYD; c += 256) {
    float val = buf[c] * scale[c >> 7];
    if (c < KEYD) qn[(size_t)t * KEYD + c]          = val;
    else          kn[(size_t)t * KEYD + (c - KEYD)] = val;
  }
}

// ---------------------------------------------------------------- delta-rule scan
// one block per value-head; thread v owns S[:, v] (128 regs).
// fused: decay, kv, delta, state update, q-dot, RMS norm, silu(z) gate.
__global__ __launch_bounds__(128) void k_scan(
    const float* __restrict__ qn, const float* __restrict__ kn,
    const float* __restrict__ vc, const float* __restrict__ gexp,
    const float* __restrict__ beta, const unsigned short* __restrict__ qkvz,
    const float* __restrict__ norm_w, unsigned short* __restrict__ core) {
  const int hv = blockIdx.x;
  const int v  = threadIdx.x;
  const int kh = hv >> 1;
  __shared__ float qs[128], ks[128], part[4];

  float S[DK];
#pragma unroll
  for (int i = 0; i < DK; i++) S[i] = 0.f;

  const float  nw   = norm_w[v];
  const int    zcol = kh * 768 + 512 + (hv & 1) * 128 + v;
  const size_t qoff = (size_t)kh * 128 + v;

  // software-pipelined preload of step 0
  float q_n  = qn[qoff];
  float k_n  = kn[qoff];
  float ge_n = gexp[hv];
  float bt_n = beta[hv];
  float vt_n = vc[(size_t)hv * 128 + v];
  float z_n  = bf2f(qkvz[zcol]);

  for (int t = 0; t < SEQ; t++) {
    qs[v] = q_n; ks[v] = k_n;
    float ge = ge_n, bt = bt_n, vt = vt_n, z = z_n;
    __syncthreads();
    if (t + 1 < SEQ) {                       // issue next step's loads early
      size_t r = (size_t)(t + 1);
      q_n  = qn[r * KEYD + qoff];
      k_n  = kn[r * KEYD + qoff];
      ge_n = gexp[r * HV + hv];
      bt_n = beta[r * HV + hv];
      vt_n = vc[r * VALD + hv * 128 + v];
      z_n  = bf2f(qkvz[r * NQKVZ + zcol]);
    }
    float kv = 0.f;
#pragma unroll
    for (int i = 0; i < DK; i++) { S[i] *= ge; kv += S[i] * ks[i]; }
    float delta = (vt - kv) * bt;
    float out = 0.f;
#pragma unroll
    for (int i = 0; i < DK; i++) { S[i] += ks[i] * delta; out += S[i] * qs[i]; }
    // cross-thread mean(out^2) over the 128 DV columns
    float o2 = out * out;
    for (int off = 16; off > 0; off >>= 1) o2 += __shfl_down(o2, off, 32);
    if ((v & 31) == 0) part[v >> 5] = o2;
    __syncthreads();
    float mean = (part[0] + part[1] + part[2] + part[3]) * (1.f / 128.f);
    float rs = rsqrtf(mean + 1e-6f);
    core[(size_t)t * VALD + hv * 128 + v] = f2bf(out * rs * nw * (z * sigm(z)));
  }
}

// ---------------------------------------------------------------- launch
extern "C" void kernel_launch(void* const* d_in, const int* in_sizes, int n_in,
                              void* d_out, int out_size, void* d_ws, size_t ws_size,
                              hipStream_t stream) {
  const float* hidden  = (const float*)d_in[0];
  const float* W_qkvz  = (const float*)d_in[1];
  const float* W_ba    = (const float*)d_in[2];
  const float* conv_w  = (const float*)d_in[3];
  const float* A_log   = (const float*)d_in[4];
  const float* dt_bias = (const float*)d_in[5];
  const float* norm_w  = (const float*)d_in[6];
  const float* W_out   = (const float*)d_in[7];
  (void)in_sizes; (void)n_in; (void)out_size; (void)ws_size;

  char*  ws  = (char*)d_ws;
  size_t off = 0;
  auto alloc = [&](size_t bytes) -> char* {
    char* p = ws + off;
    off += (bytes + 255) & ~(size_t)255;
    return p;
  };
  unsigned short* hid_bf  = (unsigned short*)alloc((size_t)SEQ * HID * 2);
  unsigned short* wqkvz_t = (unsigned short*)alloc((size_t)NQKVZ * HID * 2);
  unsigned short* wout_t  = (unsigned short*)alloc((size_t)HID * VALD * 2);
  unsigned short* qkvz_bf = (unsigned short*)alloc((size_t)SEQ * NQKVZ * 2);
  float* qn   = (float*)alloc((size_t)SEQ * KEYD * 4);
  float* kn   = (float*)alloc((size_t)SEQ * KEYD * 4);
  float* vc   = (float*)alloc((size_t)SEQ * VALD * 4);
  float* gex  = (float*)alloc((size_t)SEQ * HV * 4);
  float* bet  = (float*)alloc((size_t)SEQ * HV * 4);
  unsigned short* core_bf = (unsigned short*)alloc((size_t)SEQ * VALD * 2);

  // 1) precision conversion / tiled weight transposes (coalesced both legs)
  k_f32_to_bf16<<<(SEQ * HID + 255) / 256, 256, 0, stream>>>(hidden, hid_bf, SEQ * HID);
  {
    dim3 gt(NQKVZ / 32, HID / 32);
    k_transpose_bf16<<<gt, 256, 0, stream>>>(W_qkvz, wqkvz_t, HID, NQKVZ);
  }
  {
    dim3 gt(HID / 32, VALD / 32);
    k_transpose_bf16<<<gt, 256, 0, stream>>>(W_out, wout_t, VALD, HID);
  }

  // 2) qkvz = hidden @ W_qkvz   (WMMA bf16, f32 acc, bf16 out)
  dim3 g1(NQKVZ / 256, SEQ / 64);
  k_gemm_bf16<true><<<g1, 256, 0, stream>>>(hid_bf, wqkvz_t, qkvz_bf, SEQ, NQKVZ, HID);

  // 3) ba projection -> exp(g), beta   (tiny, fp32 exact)
  k_ba<<<SEQ, 64, 0, stream>>>(hidden, W_ba, A_log, dt_bias, gex, bet);

  // 4) causal conv(K=4) + SiLU + per-head L2 norm of q,k
  k_conv<<<SEQ, 256, 0, stream>>>(qkvz_bf, conv_w, qn, kn, vc);

  // 5) sequential delta-rule scan, fused RMS-norm + silu(z) gate
  k_scan<<<HV, DV, 0, stream>>>(qn, kn, vc, gex, bet, qkvz_bf, norm_w, core_bf);

  // 6) out = core @ W_out  (WMMA bf16, fp32 out)
  dim3 g2(HID / 256, SEQ / 64);
  k_gemm_bf16<false><<<g2, 256, 0, stream>>>(core_bf, wout_t, d_out, SEQ, HID, VALD);
}
